// GATRegression_40441412059602
// MI455X (gfx1250) — compile-verified, matrix operand order checked by reading, not run on hardware
//
#include <hip/hip_runtime.h>
#include <hip/hip_bf16.h>
#include <math.h>

#define N_NODES 10000
#define E_RAW   320000
#define E_TOT   330000
#define N_GRAPHS 64

typedef _Float16 v16h __attribute__((ext_vector_type(16)));
typedef _Float16 v8h  __attribute__((ext_vector_type(8)));
typedef float    v8f  __attribute__((ext_vector_type(8)));
typedef unsigned v4u  __attribute__((ext_vector_type(4)));
typedef int      v4i  __attribute__((ext_vector_type(4)));
typedef int      v8i  __attribute__((ext_vector_type(8)));

// ---------------------------------------------------------------------------
// TDM 2D tile load: global (row-major, f16, row stride = stride_d0 elements)
// -> LDS compacted tile [tile_d1][tile_d0].  D# per CDNA5 ISA sec. 8.
// Rows beyond tensor_d1 are zero-filled by the TDM (OOB reads return zero).
// ---------------------------------------------------------------------------
__device__ __forceinline__ void tdm_load_2d(
    unsigned lds_byte, const _Float16* gptr,
    int tensor_d0, int tensor_d1, int tile_d0, int tile_d1, int stride_d0)
{
  unsigned long long ga = (unsigned long long)(size_t)gptr;
  v4u g0;
  g0[0] = 1u;                                      // count=1, no gather
  g0[1] = lds_byte;                                // lds_addr (bytes)
  g0[2] = (unsigned)(ga & 0xFFFFFFFFu);            // global_addr[95:64]
  g0[3] = (unsigned)((ga >> 32) & 0x1FFFFFFu)      // global_addr[120:96]
        | (2u << 30);                              // type=2 ("image")
  v8i g1;
  g1[0] = 0x00010000;                              // data_size=1 (2 bytes)
  g1[1] = (int)(((unsigned)tensor_d0 & 0xFFFFu) << 16);            // dim0 lo
  g1[2] = (int)(((unsigned)tensor_d0 >> 16) |
                (((unsigned)tensor_d1 & 0xFFFFu) << 16));          // dim0 hi | dim1 lo
  g1[3] = (int)(((unsigned)tensor_d1 >> 16) |
                ((unsigned)tile_d0 << 16));                        // dim1 hi | tile_dim0
  g1[4] = (int)((unsigned)tile_d1);                                // tile_dim1 | tile_dim2=0
  g1[5] = (int)(unsigned)stride_d0;                                // dim0_stride[31:0]
  g1[6] = 0;                                                       // stride hi | dim1_stride lo
  g1[7] = 0;
  v4i z4 = {0, 0, 0, 0};
#if __clang_major__ >= 23
  v8i z8 = {0, 0, 0, 0, 0, 0, 0, 0};
  __builtin_amdgcn_tensor_load_to_lds(g0, g1, z4, z4, z8, 0);
#else
  __builtin_amdgcn_tensor_load_to_lds(g0, g1, z4, z4, 0);
#endif
}

// ---------------------------------------------------------------------------
// GEMM C[MxN] = A[MxK] * Bt[NxK]^T, f16 inputs, f32 WMMA accumulate.
// Block tile 128x64, 8 waves x (16M x 4x16N).  TDM double-buffered staging.
// ---------------------------------------------------------------------------
#define BM 128
#define BN 64
#define BK 32

__global__ __launch_bounds__(256) void gemm_f16wmma(
    const _Float16* __restrict__ A,   // [M,K] row-major f16
    const _Float16* __restrict__ Bt,  // [N,K] row-major f16 (pre-transposed B)
    float* __restrict__ C, int M, int N, int K)
{
  __shared__ __align__(16) _Float16 As[2][BM * BK];  // [m][k]
  __shared__ __align__(16) _Float16 Bs[2][BN * BK];  // [n][k]

  const int m0   = blockIdx.x * BM;
  const int n0   = blockIdx.y * BN;
  const int t    = threadIdx.x;
  const int wave = t >> 5;
  const int lane = t & 31;
  const int l16  = lane & 15;
  const int khalf = (lane < 16) ? 0 : 8;   // ISA 16-bit A/B fragment K base
  const int mrem = M - m0;
  const int nrem = N - n0;

  // prologue: wave 0 issues TDM for k-step 0 into buffer 0
  if (t < 32) {
    tdm_load_2d((unsigned)(size_t)&As[0][0], A + (size_t)m0 * K, BK, mrem, BK, BM, K);
    tdm_load_2d((unsigned)(size_t)&Bs[0][0], Bt + (size_t)n0 * K, BK, nrem, BK, BN, K);
  }

  v8f acc[4] = {};
  const int nsteps = K / BK;
  for (int s = 0; s < nsteps; ++s) {
    const int buf = s & 1;
    if (t < 32) __builtin_amdgcn_s_wait_tensorcnt(0);  // current tiles landed
    __syncthreads();                                   // visible to all waves
    // overlap: issue next k-step DMA into the other buffer
    if (s + 1 < nsteps && t < 32) {
      const int k1 = (s + 1) * BK;
      tdm_load_2d((unsigned)(size_t)&As[buf ^ 1][0], A + (size_t)m0 * K + k1,
                  BK, mrem, BK, BM, K);
      tdm_load_2d((unsigned)(size_t)&Bs[buf ^ 1][0], Bt + (size_t)n0 * K + k1,
                  BK, nrem, BK, BN, K);
    }

    const _Float16* Ab = As[buf];
    const _Float16* Bb = Bs[buf];
    // A fragment: two aligned 16B LDS reads per lane (ISA 16-bit 16x32 layout)
    v8h alo = *(const v8h*)&Ab[(wave * 16 + l16) * BK + khalf];
    v8h ahi = *(const v8h*)&Ab[(wave * 16 + l16) * BK + 16 + khalf];
    v16h afrag;
#pragma unroll
    for (int j = 0; j < 8; ++j) { afrag[j] = alo[j]; afrag[j + 8] = ahi[j]; }

#pragma unroll
    for (int nt = 0; nt < 4; ++nt) {
      v8h blo = *(const v8h*)&Bb[(nt * 16 + l16) * BK + khalf];
      v8h bhi = *(const v8h*)&Bb[(nt * 16 + l16) * BK + 16 + khalf];
      v16h bfrag;
#pragma unroll
      for (int j = 0; j < 8; ++j) { bfrag[j] = blo[j]; bfrag[j + 8] = bhi[j]; }
      acc[nt] = __builtin_amdgcn_wmma_f32_16x16x32_f16(
          false, afrag, false, bfrag, (short)0, acc[nt], false, false);
    }
    __syncthreads();  // reads of buf done before it is refilled at s+2
  }

  // C fragment: VGPR r -> M = r (lanes 0-15) / 8+r (lanes 16-31), N = lane%16
  const int rbase = m0 + wave * 16 + ((lane >> 4) & 1) * 8;
#pragma unroll
  for (int nt = 0; nt < 4; ++nt) {
    int col = n0 + nt * 16 + l16;
#pragma unroll
    for (int r = 0; r < 8; ++r) {
      int row = rbase + r;
      if (row < M) C[(size_t)row * N + col] = acc[nt][r];
    }
  }
}

// ---------------------------------------------------------------------------
// alpha_s[n,h] = dot(xl[n,h,:], a_src[h,:]); alpha_d likewise.  C = 128.
// ---------------------------------------------------------------------------
__global__ void alpha_kernel(const float* __restrict__ xl,
                             const float* __restrict__ a_src,
                             const float* __restrict__ a_dst,
                             float* __restrict__ as_out,
                             float* __restrict__ ad_out, int H)
{
  int n = blockIdx.x;
  int h = threadIdx.x >> 5;
  int lane = threadIdx.x & 31;
  const float* xp = xl + (size_t)n * H * 128 + h * 128;
  float s = 0.f, d = 0.f;
#pragma unroll
  for (int i = 0; i < 4; ++i) {
    float v = xp[lane + 32 * i];
    s += v * a_src[h * 128 + lane + 32 * i];
    d += v * a_dst[h * 128 + lane + 32 * i];
  }
  for (int off = 16; off; off >>= 1) {
    s += __shfl_down(s, off, 32);
    d += __shfl_down(d, off, 32);
  }
  if (lane == 0) { as_out[n * H + h] = s; ad_out[n * H + h] = d; }
}

// Ordered-uint mapping so float segment-max can use integer atomicMax
__device__ __forceinline__ unsigned f2o(float f) {
  unsigned u = __float_as_uint(f);
  return (u & 0x80000000u) ? ~u : (u | 0x80000000u);
}
__device__ __forceinline__ float o2f(unsigned u) {
  u = (u & 0x80000000u) ? (u & 0x7FFFFFFFu) : ~u;
  return __uint_as_float(u);
}

__device__ __forceinline__ void edge_sd(const int* __restrict__ ei, int e, int& s, int& d) {
  if (e < E_RAW) { s = ei[e]; d = ei[E_RAW + e]; }
  else           { s = d = e - E_RAW; }           // appended self-loops
}

__device__ __forceinline__ float leaky(float x) { return x >= 0.f ? x : 0.2f * x; }

__global__ void edge_max(const int* __restrict__ ei, const float* __restrict__ as,
                         const float* __restrict__ ad, unsigned* __restrict__ emax, int H)
{
  int i = blockIdx.x * blockDim.x + threadIdx.x;
  if (i >= E_TOT) return;
  int s, d; edge_sd(ei, i, s, d);
  for (int h = 0; h < H; ++h) {
    float e = leaky(as[s * H + h] + ad[d * H + h]);
    atomicMax(&emax[d * H + h], f2o(e));
  }
}

__global__ void edge_den(const int* __restrict__ ei, const float* __restrict__ as,
                         const float* __restrict__ ad, const unsigned* __restrict__ emax,
                         float* __restrict__ den, int H)
{
  int i = blockIdx.x * blockDim.x + threadIdx.x;
  if (i >= E_TOT) return;
  int s, d; edge_sd(ei, i, s, d);
  for (int h = 0; h < H; ++h) {
    float e = leaky(as[s * H + h] + ad[d * H + h]);
    atomicAdd(&den[d * H + h], expf(e - o2f(emax[d * H + h])));
  }
}

// Pass C: out[dst,h,:] += xl[src,h,:] * alpha.  One wave per (edge, head).
__global__ __launch_bounds__(256) void edge_agg(
    const int* __restrict__ ei, const float* __restrict__ as,
    const float* __restrict__ ad, const unsigned* __restrict__ emax,
    const float* __restrict__ den, const float* __restrict__ xl,
    float* __restrict__ out, int H)
{
  int wid = blockIdx.x * 8 + (threadIdx.x >> 5);
  int lane = threadIdx.x & 31;
  if (wid >= E_TOT * H) return;
  int e = wid / H, h = wid - e * H;
  int s, d; edge_sd(ei, e, s, d);
  float ev = leaky(as[s * H + h] + ad[d * H + h]);
  float alpha = expf(ev - o2f(emax[d * H + h])) / (den[d * H + h] + 1e-16f);
  const float* xp = xl + (size_t)s * H * 128 + h * 128;
  float* op = out + (size_t)d * H * 128 + h * 128;
#pragma unroll
  for (int i = 0; i < 4; ++i)
    atomicAdd(&op[lane + 32 * i], xp[lane + 32 * i] * alpha);
}

__global__ void fill_u32(unsigned* __restrict__ p, unsigned v, size_t n)
{
  size_t i  = (size_t)blockIdx.x * blockDim.x + threadIdx.x;
  size_t st = (size_t)gridDim.x * blockDim.x;
  for (; i < n; i += st) p[i] = v;
}

__global__ void f32_to_f16(const float* __restrict__ in, _Float16* __restrict__ out, size_t n)
{
  size_t i  = (size_t)blockIdx.x * blockDim.x + threadIdx.x;
  size_t st = (size_t)gridDim.x * blockDim.x;
  for (; i < n; i += st) out[i] = (_Float16)in[i];
}

// in [R][Cc] f32  ->  out [Cc][R] f16
__global__ void transpose_to_f16(const float* __restrict__ in, _Float16* __restrict__ out,
                                 int R, int Cc)
{
  int i = blockIdx.x * blockDim.x + threadIdx.x;
  if (i < R * Cc) {
    int r = i / Cc, c = i - r * Cc;
    out[(size_t)c * R + r] = (_Float16)in[i];
  }
}

// h = elu(acc + bias) -> f16 (feeds next WMMA GEMM)
__global__ void bias_elu_to_f16(const float* __restrict__ in, const float* __restrict__ b,
                                _Float16* __restrict__ out, size_t n, int cmask)
{
  size_t i  = (size_t)blockIdx.x * blockDim.x + threadIdx.x;
  size_t st = (size_t)gridDim.x * blockDim.x;
  for (; i < n; i += st) {
    float v = in[i] + b[(int)(i & (size_t)cmask)];
    out[i] = (_Float16)(v > 0.f ? v : (expf(v) - 1.f));
  }
}

// in-place f32 version (layer 2 output feeds the pooling in f32)
__global__ void bias_elu_f32(float* __restrict__ x, const float* __restrict__ b,
                             size_t n, int cmask)
{
  size_t i  = (size_t)blockIdx.x * blockDim.x + threadIdx.x;
  size_t st = (size_t)gridDim.x * blockDim.x;
  for (; i < n; i += st) {
    float v = x[i] + b[(int)(i & (size_t)cmask)];
    x[i] = v > 0.f ? v : (expf(v) - 1.f);
  }
}

__global__ __launch_bounds__(256) void pool_sum(
    const float* __restrict__ h2, const int* __restrict__ batch,
    float* __restrict__ sums, float* __restrict__ cnt)
{
  int wid = blockIdx.x * 8 + (threadIdx.x >> 5);
  int lane = threadIdx.x & 31;
  if (wid >= N_NODES) return;
  int g = batch[wid];
  const float* hp = h2 + (size_t)wid * 128;
  float* sp = sums + (size_t)g * 128;
#pragma unroll
  for (int i = 0; i < 4; ++i) atomicAdd(&sp[lane + 32 * i], hp[lane + 32 * i]);
  if (lane == 0) atomicAdd(&cnt[g], 1.f);
}

__global__ void pool_div(float* __restrict__ pooled, const float* __restrict__ sums,
                         const float* __restrict__ cnt)
{
  int i = blockIdx.x * blockDim.x + threadIdx.x;
  if (i < N_GRAPHS * 128) {
    int g = i >> 7;
    pooled[i] = sums[i] / fmaxf(cnt[g], 1.0f);
  }
}

__global__ __launch_bounds__(256) void final_head(
    const float* __restrict__ z, const float* __restrict__ lin1b,
    const float* __restrict__ lin2w, const float* __restrict__ lin2b,
    float* __restrict__ out)
{
  int wid = blockIdx.x * 8 + (threadIdx.x >> 5);
  int lane = threadIdx.x & 31;
  if (wid >= N_GRAPHS) return;
  float acc = 0.f;
#pragma unroll
  for (int i = 0; i < 4; ++i) {
    int c = lane + 32 * i;
    float v = fmaxf(z[wid * 128 + c] + lin1b[c], 0.f);
    acc += v * lin2w[c];
  }
  for (int off = 16; off; off >>= 1) acc += __shfl_down(acc, off, 32);
  if (lane == 0) out[wid] = acc + lin2b[0];
}

extern "C" void kernel_launch(void* const* d_in, const int* in_sizes, int n_in,
                              void* d_out, int out_size, void* d_ws, size_t ws_size,
                              hipStream_t stream) {
  (void)in_sizes; (void)n_in; (void)out_size; (void)ws_size;
  const float* x      = (const float*)d_in[0];
  const int*   ei     = (const int*)d_in[1];
  const int*   batch  = (const int*)d_in[2];
  const float* W1     = (const float*)d_in[3];
  const float* a_src1 = (const float*)d_in[4];
  const float* a_dst1 = (const float*)d_in[5];
  const float* b1     = (const float*)d_in[6];
  const float* W2     = (const float*)d_in[7];
  const float* a_src2 = (const float*)d_in[8];
  const float* a_dst2 = (const float*)d_in[9];
  const float* b2     = (const float*)d_in[10];
  const float* lin1w  = (const float*)d_in[11];
  const float* lin1b  = (const float*)d_in[12];
  const float* lin2w  = (const float*)d_in[13];
  const float* lin2b  = (const float*)d_in[14];
  float* out = (float*)d_out;

  // ---- workspace layout (float offsets) ----
  float* ws     = (float*)d_ws;
  float* xl1    = ws;                            // 10,240,000 f32
  float* h1     = ws + 10240000;                 // 10,240,000 f32 accumulator
  float* as1    = ws + 20480000;                 // 80,000
  float* ad1    = ws + 20560000;                 // 80,000
  unsigned* emax1 = (unsigned*)(ws + 20640000);  // 80,000
  float* den1   = ws + 20720000;                 // 80,000
  float* as2    = ws + 20800000;                 // 10,000
  float* ad2    = ws + 20810000;                 // 10,000
  unsigned* emax2 = (unsigned*)(ws + 20820000);  // 10,000
  float* den2   = ws + 20830000;                 // 10,000
  float* sums   = ws + 20840000;                 // 8,192
  float* cnt    = ws + 20848192;                 // 64
  float* pooled = ws + 20848256;                 // 8,192
  float* z      = ws + 20856448;                 // 8,192
  _Float16* xh     = (_Float16*)(ws + 20864640); // 1,280,000 halves
  _Float16* W1t    = (_Float16*)(ws + 21504640); // 131,072 halves [1024][128]
  _Float16* W2t    = (_Float16*)(ws + 21570176); // 131,072 halves [128][1024]
  _Float16* lin1wt = (_Float16*)(ws + 21635712); // 16,384 halves  [128][128]
  _Float16* pooledh= (_Float16*)(ws + 21643904); // 8,192 halves
  // aliases inside the dead xl1 region (layer-2 stage)
  float*    xl2 = ws;                            // 1,280,000 f32
  float*    h2  = ws + 1280000;                  // 1,280,000 f32
  _Float16* h1h = (_Float16*)(ws + 2560000);     // 10,240,000 halves (GEMM2 A input)

  // ---- operand conversion / transposition (once per call) ----
  f32_to_f16<<<1250, 256, 0, stream>>>(x, xh, (size_t)1280000);
  transpose_to_f16<<<512, 256, 0, stream>>>(W1, W1t, 128, 1024);
  transpose_to_f16<<<512, 256, 0, stream>>>(W2, W2t, 1024, 128);
  transpose_to_f16<<<64, 256, 0, stream>>>(lin1w, lin1wt, 128, 128);

  // ---- layer 1: GAT(128 -> 8 heads x 128) ----
  gemm_f16wmma<<<dim3(79, 16), 256, 0, stream>>>(xh, W1t, xl1, N_NODES, 1024, 128);
  alpha_kernel<<<N_NODES, 256, 0, stream>>>(xl1, a_src1, a_dst1, as1, ad1, 8);
  fill_u32<<<2048, 256, 0, stream>>>((unsigned*)h1, 0u, (size_t)10240000);
  fill_u32<<<64, 256, 0, stream>>>(emax1, 0u, (size_t)80000);
  fill_u32<<<64, 256, 0, stream>>>((unsigned*)den1, 0u, (size_t)80000);
  edge_max<<<(E_TOT + 255) / 256, 256, 0, stream>>>(ei, as1, ad1, emax1, 8);
  edge_den<<<(E_TOT + 255) / 256, 256, 0, stream>>>(ei, as1, ad1, emax1, den1, 8);
  edge_agg<<<E_TOT, 256, 0, stream>>>(ei, as1, ad1, emax1, den1, xl1, h1, 8);
  bias_elu_to_f16<<<4096, 256, 0, stream>>>(h1, b1, h1h, (size_t)10240000, 1023);

  // ---- layer 2: GAT(1024 -> 1 head x 128) ----
  gemm_f16wmma<<<dim3(79, 2), 256, 0, stream>>>(h1h, W2t, xl2, N_NODES, 128, 1024);
  alpha_kernel<<<N_NODES, 32, 0, stream>>>(xl2, a_src2, a_dst2, as2, ad2, 1);
  fill_u32<<<1024, 256, 0, stream>>>((unsigned*)h2, 0u, (size_t)1280000);
  fill_u32<<<16, 256, 0, stream>>>(emax2, 0u, (size_t)10000);
  fill_u32<<<16, 256, 0, stream>>>((unsigned*)den2, 0u, (size_t)10000);
  edge_max<<<(E_TOT + 255) / 256, 256, 0, stream>>>(ei, as2, ad2, emax2, 1);
  edge_den<<<(E_TOT + 255) / 256, 256, 0, stream>>>(ei, as2, ad2, emax2, den2, 1);
  edge_agg<<<(E_TOT + 7) / 8, 256, 0, stream>>>(ei, as2, ad2, emax2, den2, xl2, h2, 1);
  bias_elu_f32<<<512, 256, 0, stream>>>(h2, b2, (size_t)1280000, 127);

  // ---- mean pool + MLP head ----
  fill_u32<<<4, 256, 0, stream>>>((unsigned*)sums, 0u, (size_t)(N_GRAPHS * 128));
  fill_u32<<<1, 64, 0, stream>>>((unsigned*)cnt, 0u, (size_t)N_GRAPHS);
  pool_sum<<<(N_NODES + 7) / 8, 256, 0, stream>>>(h2, batch, sums, cnt);
  pool_div<<<32, 256, 0, stream>>>(pooled, sums, cnt);
  f32_to_f16<<<32, 256, 0, stream>>>(pooled, pooledh, (size_t)(N_GRAPHS * 128));
  gemm_f16wmma<<<dim3(1, 2), 256, 0, stream>>>(pooledh, lin1wt, z, N_GRAPHS, 128, 128);
  final_head<<<8, 256, 0, stream>>>(z, lin1b, lin2w, lin2b, out);
}
